// GraphAttentionLayer_60979945668752
// MI455X (gfx1250) — matrix operand order repaired and checked by reading, not compiled, not run on hardware
//
#include <hip/hip_runtime.h>
#include <hip/hip_bf16.h>

// GAT layer for MI455X (gfx1250, wave32, WMMA).
// h = X@W ; e = leakyrelu((s_i + n_j)*M) masked by adj ; softmax rows ; out = elu(P@h)
// Design: bf16 WMMA (16x16x32, f32 accum) for both GEMMs; M/adj streamed from HBM
// exactly once (537 MB -> ~23 us @ 23.3 TB/s roofline); per-block 16x8192 score
// stripe lives in 256 KB of CDNA5's 320 KB LDS as bf16.

#define N_NODES 8192
#define F_IN    512
#define F_OUT   256
#define ALPHA   0.2f
#define NEGINF  -9e15f

typedef __attribute__((ext_vector_type(16))) __bf16 v16bf;
typedef __attribute__((ext_vector_type(8)))  float  v8f;

union BfVec {
    uint4  u[2];
    v16bf  v;
    __bf16 h[16];
};

static __device__ __forceinline__ __bf16 f2bf(float x) { return (__bf16)x; }

// ---------------------------------------------------------------------------
// Kernel 1: h = X @ W  (fp32 in, bf16 WMMA, fp32 accum)
// One wave per 16x16 output tile. 8192x512x256.
// Writes H (fp32, row-major) and HT (bf16, transposed 256x8192) to workspace.
// ---------------------------------------------------------------------------
__global__ __launch_bounds__(256) void gat_gemm1(const float* __restrict__ X,
                                                 const float* __restrict__ W,
                                                 float* __restrict__ H,
                                                 __bf16* __restrict__ HT) {
    const int lane = threadIdx.x & 31;
    const int wave = threadIdx.x >> 5;
    const int tile = blockIdx.x * 8 + wave;          // 8192 tiles: 512 row x 16 col
    const int rowBase = (tile >> 4) * 16;
    const int colBase = (tile & 15) * 16;

    // A (16x32 bf16) layout: M = lane&15 ; lanes<16 hold K 0..15, lanes>=16 K 16..31
    const int mrow = rowBase + (lane & 15);
    const int kbA  = (lane < 16) ? 0 : 16;
    // B (32x16 bf16) layout: N = lane&15 ; lanes<16 hold K 0..15, lanes>=16 K 16..31
    const int nB  = colBase + (lane & 15);
    const int kbB = (lane < 16) ? 0 : 16;

    v8f acc = {};
    for (int k0 = 0; k0 < F_IN; k0 += 32) {
        BfVec a, b;
        const float4* xr = reinterpret_cast<const float4*>(X + (size_t)mrow * F_IN + k0 + kbA);
        float4 x0 = xr[0], x1 = xr[1], x2 = xr[2], x3 = xr[3];   // run [kbA..kbA+15]
        a.h[0] = f2bf(x0.x);  a.h[1] = f2bf(x0.y);  a.h[2]  = f2bf(x0.z);  a.h[3]  = f2bf(x0.w);
        a.h[4] = f2bf(x1.x);  a.h[5] = f2bf(x1.y);  a.h[6]  = f2bf(x1.z);  a.h[7]  = f2bf(x1.w);
        a.h[8] = f2bf(x2.x);  a.h[9] = f2bf(x2.y);  a.h[10] = f2bf(x2.z);  a.h[11] = f2bf(x2.w);
        a.h[12]= f2bf(x3.x);  a.h[13]= f2bf(x3.y);  a.h[14] = f2bf(x3.z);  a.h[15] = f2bf(x3.w);
#pragma unroll
        for (int t = 0; t < 16; ++t)
            b.h[t] = f2bf(W[(size_t)(k0 + kbB + t) * F_OUT + nB]);   // W is L2-resident (512 KB)

        acc = __builtin_amdgcn_wmma_f32_16x16x32_bf16(false, a.v, false, b.v,
                                                      (short)0, acc, false, false);
    }

    // C/D layout: VGPR r -> M = r (lanes 0-15) or r+8 (lanes 16-31), N = lane&15
    const int mOff = (lane >> 4) << 3;
    const int nOut = lane & 15;
#pragma unroll
    for (int r = 0; r < 8; ++r) {
        int m = r + mOff;
        float v = acc[r];
        H [(size_t)(rowBase + m) * F_OUT  + colBase + nOut] = v;
        HT[(size_t)(colBase + nOut) * N_NODES + rowBase + m] = f2bf(v);
    }
}

// ---------------------------------------------------------------------------
// Kernel 2: s[i] = h[i,:].a_self ; n[i] = h[i,:].a_neighs  (wave per row)
// ---------------------------------------------------------------------------
__global__ __launch_bounds__(256) void gat_attn(const float* __restrict__ H,
                                                const float* __restrict__ a_s,
                                                const float* __restrict__ a_n,
                                                float* __restrict__ S,
                                                float* __restrict__ Nv) {
    const int lane = threadIdx.x & 31;
    const int gi   = blockIdx.x * 8 + (threadIdx.x >> 5);
    float ss = 0.f, nn = 0.f;
    for (int c = lane; c < F_OUT; c += 32) {
        float hv = H[(size_t)gi * F_OUT + c];
        ss += hv * a_s[c];
        nn += hv * a_n[c];
    }
#pragma unroll
    for (int o = 16; o; o >>= 1) {
        ss += __shfl_xor(ss, o, 32);
        nn += __shfl_xor(nn, o, 32);
    }
    if (lane == 0) { S[gi] = ss; Nv[gi] = nn; }
}

// ---------------------------------------------------------------------------
// Kernel 3: fused scores + softmax + P@H + ELU. 16 rows per block (8 waves).
// LDS: 16*8192 bf16 = 256 KB stripe of exp-scores (CDNA5 320 KB LDS).
// M/adj read from HBM exactly once.
// ---------------------------------------------------------------------------
__global__ __launch_bounds__(256) void gat_softmax_pv(const float* __restrict__ Mw,
                                                      const float* __restrict__ ADJ,
                                                      const float* __restrict__ S,
                                                      const float* __restrict__ Nv,
                                                      const __bf16* __restrict__ HT,
                                                      float* __restrict__ OUT) {
    extern __shared__ __bf16 pbuf[];            // [16][8192]
    __shared__ float rmax[16], rsum[16];

    const int lane = threadIdx.x & 31;
    const int wave = threadIdx.x >> 5;
    const int rb   = blockIdx.x * 16;

    // ---- pass 1: e = mask(leakyrelu((s_i + n_j)*M)), track row max, stash bf16 in LDS
#pragma unroll
    for (int rr = 0; rr < 2; ++rr) {
        const int r  = wave * 2 + rr;
        const int gi = rb + r;
        const float si = S[gi];
        const float* mrow = Mw  + (size_t)gi * N_NODES;
        const float* arow = ADJ + (size_t)gi * N_NODES;
        float mx = -INFINITY;
        for (int j = lane; j < N_NODES; j += 32) {
            float e = (si + Nv[j]) * mrow[j];
            e = (e > 0.f) ? e : ALPHA * e;
            e = (arow[j] > 0.f) ? e : NEGINF;
            pbuf[r * N_NODES + j] = f2bf(e);
            mx = fmaxf(mx, e);
            __builtin_prefetch(mrow + j + 2048, 0, 1);   // global_prefetch_b8
            __builtin_prefetch(arow + j + 2048, 0, 1);
        }
#pragma unroll
        for (int o = 16; o; o >>= 1) mx = fmaxf(mx, __shfl_xor(mx, o, 32));
        if (lane == 0) rmax[r] = mx;
    }
    __syncthreads();

    // ---- pass 2: p = exp(e - max) in place (LDS), row sums
#pragma unroll
    for (int rr = 0; rr < 2; ++rr) {
        const int r = wave * 2 + rr;
        const float mx = rmax[r];
        float sum = 0.f;
        for (int j = lane; j < N_NODES; j += 32) {
            float p = __expf((float)pbuf[r * N_NODES + j] - mx);
            sum += p;
            pbuf[r * N_NODES + j] = f2bf(p);
        }
#pragma unroll
        for (int o = 16; o; o >>= 1) sum += __shfl_xor(sum, o, 32);
        if (lane == 0) rsum[r] = sum;
    }
    __syncthreads();

    // ---- pass 3: [16 x 8192] @ [8192 x 256] via bf16 WMMA; wave owns 2 col tiles
    const int rowA = lane & 15;
    const int kbA  = (lane < 16) ? 0 : 16;
    const int nb   = lane & 15;
    const int kbB  = (lane < 16) ? 0 : 16;
    const int col0 = (wave * 2 + 0) * 16;
    const int col1 = (wave * 2 + 1) * 16;
    const __bf16* bt0 = HT + (size_t)(col0 + nb) * N_NODES + kbB;
    const __bf16* bt1 = HT + (size_t)(col1 + nb) * N_NODES + kbB;

    v8f acc0 = {}, acc1 = {};
    for (int k0 = 0; k0 < N_NODES; k0 += 32) {
        BfVec a, b0, b1;
        const uint4* ap = reinterpret_cast<const uint4*>(pbuf + rowA * N_NODES + k0 + kbA);
        a.u[0] = ap[0];          // halves kbA..kbA+7       (ds_load_b128)
        a.u[1] = ap[1];          // halves kbA+8..kbA+15    (ds_load_b128)
        const uint4* bp0 = reinterpret_cast<const uint4*>(bt0 + k0);
        b0.u[0] = bp0[0]; b0.u[1] = bp0[1];     // 16 contiguous halves (global_load_b128 x2)
        const uint4* bp1 = reinterpret_cast<const uint4*>(bt1 + k0);
        b1.u[0] = bp1[0]; b1.u[1] = bp1[1];
        acc0 = __builtin_amdgcn_wmma_f32_16x16x32_bf16(false, a.v, false, b0.v,
                                                       (short)0, acc0, false, false);
        acc1 = __builtin_amdgcn_wmma_f32_16x16x32_bf16(false, a.v, false, b1.v,
                                                       (short)0, acc1, false, false);
    }

    // ---- epilogue: normalize (softmax is linear in accumulator) + ELU
    const int mOff = (lane >> 4) << 3;
#pragma unroll
    for (int r = 0; r < 8; ++r) {
        const int m = r + mOff;
        const float inv = 1.f / rsum[m];
        float x0 = acc0[r] * inv;  x0 = (x0 > 0.f) ? x0 : expm1f(x0);
        float x1 = acc1[r] * inv;  x1 = (x1 > 0.f) ? x1 : expm1f(x1);
        OUT[(size_t)(rb + m) * F_OUT + col0 + (lane & 15)] = x0;
        OUT[(size_t)(rb + m) * F_OUT + col1 + (lane & 15)] = x1;
    }
}

// ---------------------------------------------------------------------------
extern "C" void kernel_launch(void* const* d_in, const int* in_sizes, int n_in,
                              void* d_out, int out_size, void* d_ws, size_t ws_size,
                              hipStream_t stream) {
    (void)in_sizes; (void)n_in; (void)out_size; (void)ws_size;
    const float* X        = (const float*)d_in[0];   // [8192,512]
    const float* W        = (const float*)d_in[1];   // [512,256]
    const float* a_self   = (const float*)d_in[2];   // [256]
    const float* a_neighs = (const float*)d_in[3];   // [256]
    const float* ADJ      = (const float*)d_in[4];   // [8192,8192]
    const float* Mw       = (const float*)d_in[5];   // [8192,8192]
    float* OUT = (float*)d_out;                      // [8192,256]

    char* ws = (char*)d_ws;
    float*  H  = (float*)ws;                                             // 8 MB fp32
    __bf16* HT = (__bf16*)(ws + (size_t)N_NODES * F_OUT * sizeof(float)); // 4 MB bf16 (transposed)
    float*  S  = (float*)(ws + (size_t)12 * 1024 * 1024);                 // 32 KB
    float*  Nv = S + N_NODES;                                             // 32 KB

    gat_gemm1<<<1024, 256, 0, stream>>>(X, W, H, HT);
    gat_attn <<<1024, 256, 0, stream>>>(H, a_self, a_neighs, S, Nv);
    gat_softmax_pv<<<512, 256, (size_t)16 * N_NODES * sizeof(__bf16), stream>>>(
        Mw, ADJ, S, Nv, HT, OUT);
}
